// GNN_82652350644586
// MI455X (gfx1250) — matrix operand order repaired and checked by reading, not compiled, not run on hardware
//
#include <hip/hip_runtime.h>
#include <math.h>

// ---------------------------------------------------------------------------
// GNN pipeline for MI455X (gfx1250, wave32, WMMA).
//   x[50000,2048] -> maxpool2 -> LN(1024) -> (1024->512->128->64 MLP, ELU)
//   -> GCN(64->32, ELU) -> GCN(32->16) over 3.2M edges + self loops.
// Dense GEMMs: v_wmma_f32_16x16x32_f16, register-tiled 16xNT*16 per wave so
// 4 independent WMMAs issue per K-step (hides load latency, 4x A reuse).
// ---------------------------------------------------------------------------

typedef __attribute__((ext_vector_type(16))) _Float16 v16h;
typedef __attribute__((ext_vector_type(8)))  _Float16 v8h;
typedef __attribute__((ext_vector_type(8)))  float    v8f;

#define N_NODES 50000
#define N_EDGES 3200000LL
#define FEAT_IN 2048

__device__ __forceinline__ float elu_f(float v) {
    return v > 0.0f ? v : expm1f(v);
}

// ---------------------------------------------------------------------------
// 1) Fused maxpool(2:1) + LayerNorm(1024) -> f16. One block (256 thr) per node.
// ---------------------------------------------------------------------------
__global__ void pool_ln_kernel(const float* __restrict__ x,
                               const float* __restrict__ lnw,
                               const float* __restrict__ lnb,
                               _Float16* __restrict__ out) {
    const int n = blockIdx.x;
    const int t = threadIdx.x;
    const float2* xr = (const float2*)(x + (size_t)n * FEAT_IN);

    float m[4];
    float s = 0.f, s2 = 0.f;
#pragma unroll
    for (int i = 0; i < 4; ++i) {
        int j = t + 256 * i;            // output channel, coalesced
        float2 v = xr[j];               // window of 2
        m[i] = fmaxf(v.x, v.y);
        s  += m[i];
        s2 += m[i] * m[i];
    }
    // wave32 reduction
#pragma unroll
    for (int off = 16; off > 0; off >>= 1) {
        s  += __shfl_down(s,  off, 32);
        s2 += __shfl_down(s2, off, 32);
    }
    __shared__ float ss[8], ss2[8];
    __shared__ float mu, rstd;
    const int lane = t & 31, wid = t >> 5;
    if (lane == 0) { ss[wid] = s; ss2[wid] = s2; }
    __syncthreads();
    if (t == 0) {
        float S = 0.f, S2 = 0.f;
#pragma unroll
        for (int k = 0; k < 8; ++k) { S += ss[k]; S2 += ss2[k]; }
        float mean = S * (1.0f / 1024.0f);
        float var  = S2 * (1.0f / 1024.0f) - mean * mean;
        mu = mean;
        rstd = rsqrtf(var + 1e-5f);
    }
    __syncthreads();
    _Float16* orow = out + (size_t)n * 1024;
#pragma unroll
    for (int i = 0; i < 4; ++i) {
        int j = t + 256 * i;
        float h = (m[i] - mu) * rstd * lnw[j] + lnb[j];
        orow[j] = (_Float16)h;
    }
}

// ---------------------------------------------------------------------------
// 2) Pack f32 weights [K,F] row-major into WMMA B-fragment order (f16).
//    Fragment (kt,nt): lane l holds 16 halves, n = nt*16 + (l&15),
//    k = kt*32 + (l>>4)*8 + (j&7) + (j>>3)*16  (j = 0..15 contiguous in memory).
// ---------------------------------------------------------------------------
__global__ void pack_w_kernel(const float* __restrict__ W,
                              _Float16* __restrict__ out, int K, int F) {
    size_t tid = (size_t)blockIdx.x * blockDim.x + threadIdx.x;
    if (tid >= (size_t)K * F) return;
    int j  = (int)(tid & 15);
    int l  = (int)((tid >> 4) & 31);
    size_t frag = tid >> 9;                 // kt * Ntiles + nt
    int Ntiles = F >> 4;
    int nt = (int)(frag % Ntiles);
    int kt = (int)(frag / Ntiles);
    int k = kt * 32 + (l >> 4) * 8 + (j & 7) + (j >> 3) * 16;
    int n = nt * 16 + (l & 15);
    out[tid] = (_Float16)W[(size_t)k * F + n];
}

// ---------------------------------------------------------------------------
// 3) WMMA GEMM: C[M,F] = act(A[M,K] @ W + bias).
//    One wave -> 16 x (NT*16) output tile: NT accumulators, NT independent
//    WMMAs per K-step, A fragment reused NT times.
//    A layout per ISA 16-bit A-matrix 16x32: lanes 0-15 hold K {0..7,16..23},
//    lanes 16-31 hold K {8..15,24..31} for row m = lane&15.
// ---------------------------------------------------------------------------
template <int NT>
__global__ void wmma_gemm_kernel(const _Float16* __restrict__ A,
                                 const _Float16* __restrict__ Bp,
                                 const float* __restrict__ bias,
                                 _Float16* __restrict__ outH,
                                 float* __restrict__ outF,
                                 int K, int F, int elu) {
    const int lane   = threadIdx.x & 31;
    const int wave   = threadIdx.x >> 5;
    const int tileM  = blockIdx.x;
    const int Ntiles = F >> 4;
    const int tileN0 = (blockIdx.y * (blockDim.x >> 5) + wave) * NT;
    if (tileN0 >= Ntiles) return;

    const int m  = tileM * 16 + (lane & 15);
    const int kA = (lane >> 4) * 8;                 // lane-half K interleave
    const _Float16* arow = A + (size_t)m * K;
    const _Float16* bfrag[NT];
#pragma unroll
    for (int nt = 0; nt < NT; ++nt)
        bfrag[nt] = Bp + ((size_t)(tileN0 + nt) * 32 + lane) * 16;
    const size_t bstride = (size_t)Ntiles * 512;    // halves per K-step

    v8f acc[NT];
#pragma unroll
    for (int nt = 0; nt < NT; ++nt) acc[nt] = (v8f){};

    const int ksteps = K >> 5;
    for (int ks = 0; ks < ksteps; ++ks) {
        const int kb = ks * 32 + kA;
        v8h lo = *(const v8h*)(arow + kb);          // K = kb..kb+7
        v8h hi = *(const v8h*)(arow + kb + 16);     // K = kb+16..kb+23
        v16h a = __builtin_shufflevector(lo, hi, 0, 1, 2, 3, 4, 5, 6, 7,
                                                 8, 9, 10, 11, 12, 13, 14, 15);
        if (ks + 1 < ksteps)
            __builtin_prefetch(arow + kb + 32, 0, 3);   // global_prefetch_b8
        v16h b[NT];
#pragma unroll
        for (int nt = 0; nt < NT; ++nt)
            b[nt] = *(const v16h*)(bfrag[nt] + (size_t)ks * bstride);
#pragma unroll
        for (int nt = 0; nt < NT; ++nt)
            acc[nt] = __builtin_amdgcn_wmma_f32_16x16x32_f16(
                /*neg_a=*/false, a, /*neg_b=*/false, b[nt],
                /*c_mod=*/(short)0, acc[nt],
                /*reuse_a=*/false, /*reuse_b=*/false);
    }

    // D layout: VGPR r -> M = r + 8*(lane>=16), N = lane&15
    const int mbase = tileM * 16 + (lane >> 4) * 8;
#pragma unroll
    for (int nt = 0; nt < NT; ++nt) {
        const int n = (tileN0 + nt) * 16 + (lane & 15);
        const float bv = bias ? bias[n] : 0.0f;
#pragma unroll
        for (int r = 0; r < 8; ++r) {
            float v = acc[nt][r] + bv;
            if (elu) v = elu_f(v);
            size_t idx = (size_t)(mbase + r) * F + n;
            if (outH) outH[idx] = (_Float16)v;
            if (outF) outF[idx] = v;
        }
    }
}

// ---------------------------------------------------------------------------
// 4) GCN helpers
// ---------------------------------------------------------------------------
__global__ void fill_kernel(float* p, float v, int n) {
    int i = blockIdx.x * blockDim.x + threadIdx.x;
    if (i < n) p[i] = v;
}

__global__ void degree_kernel(const long long* __restrict__ col,
                              float* __restrict__ deg, long long E) {
    for (long long e = (long long)blockIdx.x * blockDim.x + threadIdx.x; e < E;
         e += (long long)gridDim.x * blockDim.x)
        atomicAdd(&deg[col[e]], 1.0f);
}

__global__ void rsqrt_kernel(const float* __restrict__ deg,
                             float* __restrict__ dis, int n) {
    int i = blockIdx.x * blockDim.x + threadIdx.x;
    if (i < n) dis[i] = rsqrtf(deg[i]);
}

// out[col] += dis[row]*dis[col] * xw[row]  (F = 32 or 16; logF = log2(F))
__global__ void gcn_agg_kernel(const float* __restrict__ xw,
                               const long long* __restrict__ row,
                               const long long* __restrict__ col,
                               const float* __restrict__ dis,
                               float* __restrict__ out,
                               long long E, int F, int logF) {
    size_t total = (size_t)E << logF;
    for (size_t t = (size_t)blockIdx.x * blockDim.x + threadIdx.x; t < total;
         t += (size_t)gridDim.x * blockDim.x) {
        size_t e = t >> logF;
        int f = (int)(t & (F - 1));
        long long r = row[e], c = col[e];
        float nm = dis[r] * dis[c];
        atomicAdd(&out[((size_t)c << logF) + f], nm * xw[((size_t)r << logF) + f]);
    }
}

// v = agg + dis^2 * xw (self loop) + bias; optional ELU; f16 and/or f32 out.
__global__ void gcn_finish_kernel(const float* __restrict__ agg,
                                  const float* __restrict__ xw,
                                  const float* __restrict__ dis,
                                  const float* __restrict__ bias,
                                  _Float16* __restrict__ outH,
                                  float* __restrict__ outF,
                                  int N, int F, int logF, int elu) {
    size_t total = (size_t)N << logF;
    size_t t = (size_t)blockIdx.x * blockDim.x + threadIdx.x;
    if (t >= total) return;
    int i = (int)(t >> logF);
    int f = (int)(t & (F - 1));
    float d = dis[i];
    float v = agg[t] + d * d * xw[t] + bias[f];
    if (elu) v = elu_f(v);
    if (outH) outH[t] = (_Float16)v;
    if (outF) outF[t] = v;
}

// ---------------------------------------------------------------------------
// Launcher
// ---------------------------------------------------------------------------
static void launch_gemm(const _Float16* A, const _Float16* Bp, const float* bias,
                        _Float16* outH, float* outF, int K, int F, int elu,
                        hipStream_t stream) {
    int Ntiles = F >> 4;
    if (Ntiles >= 4) {
        int ngroups = Ntiles / 4;                 // 16x64 tile per wave
        int wpb = ngroups < 4 ? ngroups : 4;
        dim3 grid(N_NODES / 16, (ngroups + wpb - 1) / wpb);
        wmma_gemm_kernel<4><<<grid, 32 * wpb, 0, stream>>>(A, Bp, bias, outH,
                                                           outF, K, F, elu);
    } else if (Ntiles == 2) {
        dim3 grid(N_NODES / 16, 1);
        wmma_gemm_kernel<2><<<grid, 32, 0, stream>>>(A, Bp, bias, outH, outF,
                                                     K, F, elu);
    } else {
        dim3 grid(N_NODES / 16, 1);
        wmma_gemm_kernel<1><<<grid, 32, 0, stream>>>(A, Bp, bias, outH, outF,
                                                     K, F, elu);
    }
}

extern "C" void kernel_launch(void* const* d_in, const int* in_sizes, int n_in,
                              void* d_out, int out_size, void* d_ws, size_t ws_size,
                              hipStream_t stream) {
    const float*     x    = (const float*)d_in[0];
    const long long* ei   = (const long long*)d_in[1];   // int64 [2, E]
    const float*     lnw  = (const float*)d_in[2];
    const float*     lnb  = (const float*)d_in[3];
    const float*     w0   = (const float*)d_in[4];
    const float*     b0   = (const float*)d_in[5];
    const float*     w1   = (const float*)d_in[6];
    const float*     b1   = (const float*)d_in[7];
    const float*     w2   = (const float*)d_in[8];
    const float*     b2   = (const float*)d_in[9];
    const float*     g1w  = (const float*)d_in[10];
    const float*     g1b  = (const float*)d_in[11];
    const float*     g2w  = (const float*)d_in[12];
    const float*     g2b  = (const float*)d_in[13];

    const long long E = (long long)in_sizes[1] / 2;
    const long long* row = ei;
    const long long* col = ei + E;

    char* ws = (char*)d_ws;
    // Workspace layout (bytes). Post-MLP small buffers alias the pooled f16
    // region (dead after GEMM1).
    _Float16* H    = (_Float16*)(ws + 0);                    // 102,400,000
    _Float16* h1   = (_Float16*)(ws + 102400000);            //  51,200,000
    _Float16* h2   = (_Float16*)(ws + 0);                    //  12,800,000 (alias)
    _Float16* h3   = (_Float16*)(ws + 12800000);             //   6,400,000
    float*    xw1  = (float*)   (ws + 19200000);             //   6,400,000
    float*    agg1 = (float*)   (ws + 25600000);             //   6,400,000
    _Float16* h4   = (_Float16*)(ws + 32000000);             //   3,200,000
    float*    xw2  = (float*)   (ws + 35200000);             //   3,200,000
    float*    deg  = (float*)   (ws + 153600000);            //     200,000
    float*    dis  = (float*)   (ws + 153800000);            //     200,000
    _Float16* w0p  = (_Float16*)(ws + 154000000);            //   1,048,576
    _Float16* w1p  = (_Float16*)(ws + 155048576);            //     131,072
    _Float16* w2p  = (_Float16*)(ws + 155179648);            //      16,384
    _Float16* g1p  = (_Float16*)(ws + 155196032);            //       4,096
    _Float16* g2p  = (_Float16*)(ws + 155200128);            //       1,024

    // --- pack weights to WMMA fragment order (f16) ---
    pack_w_kernel<<<(1024 * 512) / 256, 256, 0, stream>>>(w0, w0p, 1024, 512);
    pack_w_kernel<<<(512 * 128) / 256, 256, 0, stream>>>(w1, w1p, 512, 128);
    pack_w_kernel<<<(128 * 64) / 256, 256, 0, stream>>>(w2, w2p, 128, 64);
    pack_w_kernel<<<(64 * 32) / 256, 256, 0, stream>>>(g1w, g1p, 64, 32);
    pack_w_kernel<<<(32 * 16) / 256, 256, 0, stream>>>(g2w, g2p, 32, 16);

    // --- pool + layernorm -> f16 ---
    pool_ln_kernel<<<N_NODES, 256, 0, stream>>>(x, lnw, lnb, H);

    // --- MLP (WMMA) ---
    launch_gemm(H,  w0p, b0, h1, nullptr, 1024, 512, /*elu=*/1, stream);
    launch_gemm(h1, w1p, b1, h2, nullptr, 512, 128, 1, stream);
    launch_gemm(h2, w2p, b2, h3, nullptr, 128, 64, 1, stream);

    // --- degrees / normalization (self loop -> init deg = 1) ---
    fill_kernel<<<(N_NODES + 255) / 256, 256, 0, stream>>>(deg, 1.0f, N_NODES);
    degree_kernel<<<4096, 256, 0, stream>>>(col, deg, E);
    rsqrt_kernel<<<(N_NODES + 255) / 256, 256, 0, stream>>>(deg, dis, N_NODES);

    // --- GCN conv 1: 64 -> 32, ELU ---
    launch_gemm(h3, g1p, nullptr, nullptr, xw1, 64, 32, 0, stream);
    hipMemsetAsync(agg1, 0, (size_t)N_NODES * 32 * sizeof(float), stream);
    {
        size_t total = (size_t)E * 32;
        gcn_agg_kernel<<<(unsigned)((total + 255) / 256), 256, 0, stream>>>(
            xw1, row, col, dis, agg1, E, 32, 5);
    }
    gcn_finish_kernel<<<(N_NODES * 32 + 255) / 256, 256, 0, stream>>>(
        agg1, xw1, dis, g1b, h4, nullptr, N_NODES, 32, 5, /*elu=*/1);

    // --- GCN conv 2: 32 -> 16, aggregate straight into d_out ---
    launch_gemm(h4, g2p, nullptr, nullptr, xw2, 32, 16, 0, stream);
    float* outF = (float*)d_out;
    hipMemsetAsync(outF, 0, (size_t)N_NODES * 16 * sizeof(float), stream);
    {
        size_t total = (size_t)E * 16;
        gcn_agg_kernel<<<(unsigned)((total + 255) / 256), 256, 0, stream>>>(
            xw2, row, col, dis, outF, E, 16, 4);
    }
    gcn_finish_kernel<<<(N_NODES * 16 + 255) / 256, 256, 0, stream>>>(
        outF, xw2, dis, g2b, nullptr, outF, N_NODES, 16, 4, /*elu=*/0);
}
